// EdgeDecoder_41059887350179
// MI455X (gfx1250) — matrix operand order repaired and checked by reading, compiled, not run on hardware
//
#include <hip/hip_runtime.h>
#include <stdint.h>

typedef __attribute__((ext_vector_type(16))) _Float16 v16h;
typedef __attribute__((ext_vector_type(8)))  float    v8f;

#define HDIM 256       // hidden channels
#define KC_COUNT 16    // 512 / 32
#define CHUNK_DW 4096  // dwords per K-chunk of packed W1: 16 nt * 32 lanes * 8 dw

// ---------------------------------------------------------------------------
// Prep: convert W1 [512,256] f32 -> f16, pre-swizzled into CDNA5 WMMA
// B-operand (32x16 f16) fragments in this code's contiguous k order, one contiguous 16KB chunk per
// K-chunk so the main kernel can bulk-copy chunks to LDS.
// dword index = ((kc*16 + nt)*32 + lane)*8 + v
//   lane<16 : col n = nt*16+lane,     K = kc*32 + {2v, 2v+1}
//   lane>=16: col n = nt*16+lane-16,  K = kc*32 + 16 + {2v, 2v+1}
// ---------------------------------------------------------------------------
__global__ __launch_bounds__(256) void prep_w1_kernel(
    const float* __restrict__ W1, uint32_t* __restrict__ w1p) {
  int gid  = blockIdx.x * 256 + threadIdx.x;  // 0 .. 65535
  int v    = gid & 7;
  int lane = (gid >> 3) & 31;
  int nt   = (gid >> 8) & 15;
  int kc   = gid >> 12;
  int n  = nt * 16 + (lane & 15);
  int k0 = kc * 32 + ((lane & 16) ? 16 : 0) + 2 * v;
  union { _Float16 h[2]; uint32_t u; } p;
  p.h[0] = (_Float16)W1[(k0 + 0) * HDIM + n];
  p.h[1] = (_Float16)W1[(k0 + 1) * HDIM + n];
  w1p[gid] = p.u;
}

// gather 32 K-values of one row, f32 -> f16, in the contiguous k order w1p also uses:
// lanes 0-15 hold K {0..15}, lanes 16-31 hold K {16..31}.
__device__ __forceinline__ v16h load_convert_a(const float* srcA, int hi) {
  const float4* g = (const float4*)srcA + hi * 4;
  float4 f0 = g[0], f1 = g[1], f2 = g[2], f3 = g[3];
  v16h a;
  a[0]  = (_Float16)f0.x; a[1]  = (_Float16)f0.y;
  a[2]  = (_Float16)f0.z; a[3]  = (_Float16)f0.w;
  a[4]  = (_Float16)f1.x; a[5]  = (_Float16)f1.y;
  a[6]  = (_Float16)f1.z; a[7]  = (_Float16)f1.w;
  a[8]  = (_Float16)f2.x; a[9]  = (_Float16)f2.y;
  a[10] = (_Float16)f2.z; a[11] = (_Float16)f2.w;
  a[12] = (_Float16)f3.x; a[13] = (_Float16)f3.y;
  a[14] = (_Float16)f3.z; a[15] = (_Float16)f3.w;
  return a;
}

// ---------------------------------------------------------------------------
// Main: fused gather + [E,512]x[512,256] GEMM (f16 WMMA, f32 acc) + bias +
// relu + [256]->1 dot + bias.
// WG = 256 threads = 8 waves = 4 wave-pairs. Pair p owns 32 rows
// (blk*128 + p*32 ..); within a pair, wave nh=0 computes N-tiles 0-7 and
// nh=1 computes N-tiles 8-15. Each B fragment feeds 2 WMMAs (two M-tiles),
// halving LDS read traffic per WMMA.
// ---------------------------------------------------------------------------
__global__ __launch_bounds__(256) void edge_decoder_kernel(
    const float* __restrict__ user_emb, const float* __restrict__ movie_emb,
    const int*   __restrict__ eidx,     const float* __restrict__ b1,
    const float* __restrict__ W2,       const float* __restrict__ b2,
    const uint32_t* __restrict__ w1p,   float* __restrict__ out, int E) {
  __shared__ uint32_t ldsB[2][CHUNK_DW];  // double-buffered K-chunk of W1 (32 KB)
  __shared__ float    red[2][128];        // per-N-half row partials

  const int tid  = threadIdx.x;
  const int wave = tid >> 5;
  const int lane = tid & 31;
  const int lm   = lane & 15;
  const int hi   = lane >> 4;   // 0: lanes 0-15, 1: lanes 16-31
  const int pair = wave >> 1;   // 0..3
  const int nh   = wave & 1;    // N-half: tiles nh*8 .. nh*8+7

  const int rp = blockIdx.x * 128 + pair * 32;  // pair's 32-row base
  int r0  = rp + lm;
  int r1  = rp + 16 + lm;
  int rc0 = (r0 < E) ? r0 : (E - 1);  // clamp: WMMA requires full EXEC
  int rc1 = (r1 < E) ? r1 : (E - 1);
  const float* pu0 = user_emb  + (long)eidx[rc0]     * HDIM;
  const float* pm0 = movie_emb + (long)eidx[E + rc0] * HDIM;
  const float* pu1 = user_emb  + (long)eidx[rc1]     * HDIM;
  const float* pm1 = movie_emb + (long)eidx[E + rc1] * HDIM;

  // ---- stage chunk 0 into buffer 0
  {
    const int4* src = (const int4*)w1p;
    int4*       dst = (int4*)&ldsB[0][0];
#pragma unroll
    for (int i = 0; i < 4; ++i) dst[tid + i * 256] = src[tid + i * 256];
  }
  __syncthreads();

  v8f zero = {0.f, 0.f, 0.f, 0.f, 0.f, 0.f, 0.f, 0.f};
  v8f acc0[8], acc1[8];
#pragma unroll
  for (int i = 0; i < 8; ++i) { acc0[i] = zero; acc1[i] = zero; }

  const int ntbase = nh * 8;
  union BU { int4 q[2]; v16h h; };

  for (int kc = 0; kc < KC_COUNT; ++kc) {
    const uint32_t* cur = &ldsB[kc & 1][0];

    // ---- prefetch next W1 chunk into registers (lands during compute)
    int4 st0, st1, st2, st3;
    if (kc + 1 < KC_COUNT) {
      const int4* src = (const int4*)(w1p + (kc + 1) * CHUNK_DW);
      st0 = src[tid];       st1 = src[tid + 256];
      st2 = src[tid + 512]; st3 = src[tid + 768];
    }

    // ---- two A fragments (rows rp.. and rp+16..) for this K-chunk
    const float* s0 = (kc < 8) ? (pu0 + kc * 32) : (pm0 + (kc - 8) * 32);
    const float* s1 = (kc < 8) ? (pu1 + kc * 32) : (pm1 + (kc - 8) * 32);
    v16h a0 = load_convert_a(s0, hi);
    v16h a1 = load_convert_a(s1, hi);

    // ---- 8 B fragments, pipelined one ahead; each feeds 2 WMMAs
    BU bq[2];
    {
      const int4* bp = (const int4*)(cur + (ntbase * 32 + lane) * 8);
      bq[0].q[0] = bp[0]; bq[0].q[1] = bp[1];
    }
#pragma unroll
    for (int i = 0; i < 8; ++i) {
      if (i + 1 < 8) {
        const int4* bp = (const int4*)(cur + ((ntbase + i + 1) * 32 + lane) * 8);
        bq[(i + 1) & 1].q[0] = bp[0];
        bq[(i + 1) & 1].q[1] = bp[1];
      }
      acc0[i] = __builtin_amdgcn_wmma_f32_16x16x32_f16(
          false, a0, false, bq[i & 1].h, (short)0, acc0[i], false, false);
      acc1[i] = __builtin_amdgcn_wmma_f32_16x16x32_f16(
          false, a1, false, bq[i & 1].h, (short)0, acc1[i], false, false);
    }

    // ---- commit prefetched chunk to the other LDS buffer
    if (kc + 1 < KC_COUNT) {
      int4* dst = (int4*)&ldsB[(kc + 1) & 1][0];
      dst[tid]       = st0; dst[tid + 256] = st1;
      dst[tid + 512] = st2; dst[tid + 768] = st3;
    }
    __syncthreads();
  }

  // ---- fused layer 2: relu(acc + b1) . W2 over this wave's 128 N-columns.
  // C/D layout: lane = column n (mod 16); VGPR j = row j (lanes 0-15) or
  // row j+8 (lanes 16-31).
  float rs0[8], rs1[8];
#pragma unroll
  for (int j = 0; j < 8; ++j) { rs0[j] = 0.f; rs1[j] = 0.f; }
#pragma unroll
  for (int i = 0; i < 8; ++i) {
    int   n  = (ntbase + i) * 16 + lm;
    float bb = b1[n];
    float ww = W2[n];
#pragma unroll
    for (int j = 0; j < 8; ++j) {
      float t0 = acc0[i][j] + bb;
      float t1 = acc1[i][j] + bb;
      t0 = (t0 > 0.f) ? t0 : 0.f;
      t1 = (t1 > 0.f) ? t1 : 0.f;
      rs0[j] += t0 * ww;
      rs1[j] += t1 * ww;
    }
  }
  // reduce across the 16 N-lanes (masks stay within each EXEC half),
  // then deposit per-(row, N-half) partials in LDS — exactly one writer each.
#pragma unroll
  for (int j = 0; j < 8; ++j) {
    float s0 = rs0[j], s1 = rs1[j];
    s0 += __shfl_xor(s0, 1); s1 += __shfl_xor(s1, 1);
    s0 += __shfl_xor(s0, 2); s1 += __shfl_xor(s1, 2);
    s0 += __shfl_xor(s0, 4); s1 += __shfl_xor(s1, 4);
    s0 += __shfl_xor(s0, 8); s1 += __shfl_xor(s1, 8);
    if (lm == 0) {
      red[nh][pair * 32 + j + hi * 8]      = s0;
      red[nh][pair * 32 + 16 + j + hi * 8] = s1;
    }
  }
  __syncthreads();
  if (tid < 128) {
    int row = blockIdx.x * 128 + tid;
    if (row < E) out[row] = red[0][tid] + red[1][tid] + b2[0];
  }
}

extern "C" void kernel_launch(void* const* d_in, const int* in_sizes, int n_in,
                              void* d_out, int out_size, void* d_ws, size_t ws_size,
                              hipStream_t stream) {
  const float* user_emb  = (const float*)d_in[0];
  const float* movie_emb = (const float*)d_in[1];
  const int*   eidx      = (const int*)d_in[2];
  const float* W1        = (const float*)d_in[3];
  const float* b1        = (const float*)d_in[4];
  const float* W2        = (const float*)d_in[5];
  const float* b2        = (const float*)d_in[6];
  float*       out       = (float*)d_out;

  const int E = in_sizes[2] / 2;  // edge_label_index is [2, E]
  uint32_t* w1p = (uint32_t*)d_ws;  // 256 KB packed f16 W1

  prep_w1_kernel<<<256, 256, 0, stream>>>(W1, w1p);

  int grid = (E + 127) / 128;
  edge_decoder_kernel<<<grid, 256, 0, stream>>>(
      user_emb, movie_emb, eidx, b1, W2, b2, w1p, out, E);
}
